// SOTUEncoder_79937931313417
// MI455X (gfx1250) — matrix operand order repaired and checked by reading, not compiled, not run on hardware
//
#include <hip/hip_runtime.h>

#define N_TAXON 100000
#define N_SOTU  200000
#define D_FEAT  128
#define OUT_F   64
#define E_HP    200000
#define E_RH    2000000

typedef __attribute__((ext_vector_type(16))) __bf16 v16bf;
typedef __attribute__((ext_vector_type(8)))  float  v8f;

union Frag { uint4 q[2]; v16bf v; };

static __device__ __forceinline__ unsigned short f2bf(float f) {
    unsigned u = __float_as_uint(f);
    u += 0x7FFFu + ((u >> 16) & 1u);          // round-to-nearest-even
    return (unsigned short)(u >> 16);
}
static __device__ __forceinline__ float bf2f(unsigned short h) {
    return __uint_as_float(((unsigned)h) << 16);
}

// ---------------------------------------------------------------- elementwise
__global__ void k_zero(float* __restrict__ p, long n) {
    long i = (long)blockIdx.x * blockDim.x + threadIdx.x;
    if (i < n) p[i] = 0.0f;
}

__global__ void k_f32_to_bf16(const float* __restrict__ x,
                              unsigned short* __restrict__ o, long n) {
    long i = (long)blockIdx.x * blockDim.x + threadIdx.x;
    if (i < n) o[i] = f2bf(x[i]);
}

// Pack W (K=128 rows x N cols, row-major f32) into WMMA B-fragment order:
// per (kb,nt) tile of 32x16, lane l holds column n0+l%16, K = kb*32+16*(l/16)+e,
// 16 contiguous bf16 per lane -> two b128 loads at GEMM time.
__global__ void k_pack_w(const float* __restrict__ W,
                         unsigned short* __restrict__ out, int N, int total) {
    int tid = blockIdx.x * blockDim.x + threadIdx.x;
    if (tid >= total) return;
    int e    = tid & 15;
    int lane = (tid >> 4) & 31;
    int tile = tid >> 9;
    int Ntiles = N >> 4;
    int nb = tile % Ntiles, kb = tile / Ntiles;
    int n = nb * 16 + (lane & 15);
    int k = kb * 32 + (lane >> 4) * 16 + e;
    out[tid] = f2bf(W[k * N + n]);
}

// mean = sum / max(cnt,1), stored bf16
__global__ void k_mean_bf16(const float* __restrict__ sum,
                            const float* __restrict__ cnt,
                            unsigned short* __restrict__ o, int nrows) {
    long idx = (long)blockIdx.x * blockDim.x + threadIdx.x;
    if (idx >= (long)nrows * D_FEAT) return;
    int i = (int)(idx >> 7);
    float r = 1.0f / fmaxf(cnt[i], 1.0f);
    o[idx] = f2bf(sum[idx] * r);
}

// ---------------------------------------------------------------- scatter-add
// one wave per edge; lane handles features [4*lane, 4*lane+4)
__global__ void k_scatter(const unsigned short* __restrict__ X,
                          const int* __restrict__ src,
                          const int* __restrict__ dst,
                          float* __restrict__ sum,
                          float* __restrict__ cnt,
                          int E, int addcnt) {
    int wid  = (blockIdx.x * blockDim.x + threadIdx.x) >> 5;
    int lane = threadIdx.x & 31;
    if (wid >= E) return;
    int s = src[wid], t = dst[wid];
    uint2 d = *(const uint2*)(X + (size_t)s * D_FEAT + lane * 4);
    float* base = sum + (size_t)t * D_FEAT + lane * 4;
    __hip_atomic_fetch_add(base + 0, bf2f((unsigned short)(d.x & 0xFFFF)),
                           __ATOMIC_RELAXED, __HIP_MEMORY_SCOPE_AGENT);
    __hip_atomic_fetch_add(base + 1, bf2f((unsigned short)(d.x >> 16)),
                           __ATOMIC_RELAXED, __HIP_MEMORY_SCOPE_AGENT);
    __hip_atomic_fetch_add(base + 2, bf2f((unsigned short)(d.y & 0xFFFF)),
                           __ATOMIC_RELAXED, __HIP_MEMORY_SCOPE_AGENT);
    __hip_atomic_fetch_add(base + 3, bf2f((unsigned short)(d.y >> 16)),
                           __ATOMIC_RELAXED, __HIP_MEMORY_SCOPE_AGENT);
    if (addcnt && lane == 0)
        __hip_atomic_fetch_add(cnt + t, 1.0f,
                               __ATOMIC_RELAXED, __HIP_MEMORY_SCOPE_AGENT);
}

// ---------------------------------------------------------------- SAGE GEMM
// out[16 rows x 128] per wave: relu(A1@Wl + A2@Wr + b), A row-major bf16 (ld=128).
// Packed weights staged in LDS once per block (8 waves share one copy).
__global__ void __launch_bounds__(256)
k_sage_gemm(const unsigned short* __restrict__ A1,
            const unsigned short* __restrict__ A2,
            const unsigned short* __restrict__ Wl,
            const unsigned short* __restrict__ Wr,
            const float* __restrict__ bias,
            unsigned short* __restrict__ out, int n_rows) {
    __shared__ unsigned short sW[2 * 16384];      // 64 KB: Wl | Wr fragment-packed
    for (int i = threadIdx.x; i < 4096; i += blockDim.x) {
        ((uint4*)sW)[i] = (i < 2048) ? ((const uint4*)Wl)[i]
                                     : ((const uint4*)Wr)[i - 2048];
    }
    __syncthreads();

    int wid  = (blockIdx.x * blockDim.x + threadIdx.x) >> 5;
    int lane = threadIdx.x & 31;
    int row0 = wid * 16;
    if (row0 >= n_rows) return;
    int m = lane & 15, half = lane >> 4;

    v8f acc[8];
#pragma unroll
    for (int nt = 0; nt < 8; ++nt) {
        float bv = bias[nt * 16 + m];
#pragma unroll
        for (int r = 0; r < 8; ++r) acc[nt][r] = bv;
    }

    const unsigned short* rowA1 = A1 + (size_t)(row0 + m) * D_FEAT;
    const unsigned short* rowA2 = A2 + (size_t)(row0 + m) * D_FEAT;
    const unsigned short* sWl = sW;
    const unsigned short* sWr = sW + 16384;

#pragma unroll
    for (int kb = 0; kb < 4; ++kb) {
        Frag a;
        a.q[0] = *(const uint4*)(rowA1 + kb * 32 + 16 * half);
        a.q[1] = *(const uint4*)(rowA1 + kb * 32 + 16 * half + 8);
#pragma unroll
        for (int nt = 0; nt < 8; ++nt) {
            Frag b;
            const uint4* bp = (const uint4*)(sWl + (kb * 8 + nt) * 512 + lane * 16);
            b.q[0] = bp[0]; b.q[1] = bp[1];
            acc[nt] = __builtin_amdgcn_wmma_f32_16x16x32_bf16(
                false, a.v, false, b.v, (short)0, acc[nt], false, false);
        }
    }
#pragma unroll
    for (int kb = 0; kb < 4; ++kb) {
        Frag a;
        a.q[0] = *(const uint4*)(rowA2 + kb * 32 + 16 * half);
        a.q[1] = *(const uint4*)(rowA2 + kb * 32 + 16 * half + 8);
#pragma unroll
        for (int nt = 0; nt < 8; ++nt) {
            Frag b;
            const uint4* bp = (const uint4*)(sWr + (kb * 8 + nt) * 512 + lane * 16);
            b.q[0] = bp[0]; b.q[1] = bp[1];
            acc[nt] = __builtin_amdgcn_wmma_f32_16x16x32_bf16(
                false, a.v, false, b.v, (short)0, acc[nt], false, false);
        }
    }

#pragma unroll
    for (int nt = 0; nt < 8; ++nt)
#pragma unroll
        for (int r = 0; r < 8; ++r) {
            float v = acc[nt][r];
            v = v > 0.0f ? v : 0.0f;                       // relu
            out[(size_t)(row0 + r + 8 * half) * D_FEAT + nt * 16 + m] = f2bf(v);
        }
}

// final: d_out[16 x 64] per wave = A @ Wlin + blin (f32 out, no relu)
__global__ void __launch_bounds__(256)
k_final(const unsigned short* __restrict__ A,
        const unsigned short* __restrict__ Wp,
        const float* __restrict__ bias,
        float* __restrict__ out, int n_rows) {
    __shared__ unsigned short sW[8192];            // 16 KB packed Wlin
    for (int i = threadIdx.x; i < 1024; i += blockDim.x)
        ((uint4*)sW)[i] = ((const uint4*)Wp)[i];
    __syncthreads();

    int wid  = (blockIdx.x * blockDim.x + threadIdx.x) >> 5;
    int lane = threadIdx.x & 31;
    int row0 = wid * 16;
    if (row0 >= n_rows) return;
    int m = lane & 15, half = lane >> 4;

    v8f acc[4];
#pragma unroll
    for (int nt = 0; nt < 4; ++nt) {
        float bv = bias[nt * 16 + m];
#pragma unroll
        for (int r = 0; r < 8; ++r) acc[nt][r] = bv;
    }

    const unsigned short* rowA = A + (size_t)(row0 + m) * D_FEAT;
#pragma unroll
    for (int kb = 0; kb < 4; ++kb) {
        Frag a;
        a.q[0] = *(const uint4*)(rowA + kb * 32 + 16 * half);
        a.q[1] = *(const uint4*)(rowA + kb * 32 + 16 * half + 8);
#pragma unroll
        for (int nt = 0; nt < 4; ++nt) {
            Frag b;
            const uint4* bp = (const uint4*)(sW + (kb * 4 + nt) * 512 + lane * 16);
            b.q[0] = bp[0]; b.q[1] = bp[1];
            acc[nt] = __builtin_amdgcn_wmma_f32_16x16x32_bf16(
                false, a.v, false, b.v, (short)0, acc[nt], false, false);
        }
    }
#pragma unroll
    for (int nt = 0; nt < 4; ++nt)
#pragma unroll
        for (int r = 0; r < 8; ++r)
            out[(size_t)(row0 + r + 8 * half) * OUT_F + nt * 16 + m] = acc[nt][r];
}

// ---------------------------------------------------------------- launch
extern "C" void kernel_launch(void* const* d_in, const int* in_sizes, int n_in,
                              void* d_out, int out_size, void* d_ws, size_t ws_size,
                              hipStream_t stream) {
    const float* x_taxon = (const float*)d_in[0];
    const float* x_sotu  = (const float*)d_in[1];
    const int* hp_src = (const int*)d_in[2];
    const int* hp_dst = (const int*)d_in[3];
    const int* rh_src = (const int*)d_in[4];
    const int* rh_dst = (const int*)d_in[5];
    const float* W1l = (const float*)d_in[6];
    const float* W1r = (const float*)d_in[7];
    const float* b1  = (const float*)d_in[8];
    const float* W2l = (const float*)d_in[9];
    const float* W2r = (const float*)d_in[10];
    const float* b2  = (const float*)d_in[11];
    const float* W3l = (const float*)d_in[12];
    const float* W3r = (const float*)d_in[13];
    const float* b3  = (const float*)d_in[14];
    const float* Wlin = (const float*)d_in[15];
    const float* blin = (const float*)d_in[16];
    float* outp = (float*)d_out;

    char* ws = (char*)d_ws;
    size_t off = 0;
    float* sum_taxon = (float*)(ws + off); off += (size_t)N_TAXON * D_FEAT * 4;   // 51.2 MB
    float* cnt_taxon = (float*)(ws + off); off += (size_t)N_TAXON * 4;            // 0.4 MB
    float* sum_sotu  = (float*)(ws + off); off += (size_t)N_SOTU * D_FEAT * 4;    // 102.4 MB
    float* cnt_sotu  = (float*)(ws + off); off += (size_t)N_SOTU * 4;             // 0.8 MB
    unsigned short* xt_bf   = (unsigned short*)(ws + off); off += (size_t)N_TAXON * D_FEAT * 2;
    unsigned short* xs_bf   = (unsigned short*)(ws + off); off += (size_t)N_SOTU  * D_FEAT * 2;
    unsigned short* mean_t  = (unsigned short*)(ws + off); off += (size_t)N_TAXON * D_FEAT * 2;
    unsigned short* mean_s  = (unsigned short*)(ws + off); off += (size_t)N_SOTU  * D_FEAT * 2;
    unsigned short* taxon_h = (unsigned short*)(ws + off); off += (size_t)N_TAXON * D_FEAT * 2;
    unsigned short* sotu_h  = (unsigned short*)(ws + off); off += (size_t)N_SOTU  * D_FEAT * 2;
    unsigned short* w1l_p = (unsigned short*)(ws + off); off += 128 * 128 * 2;
    unsigned short* w1r_p = (unsigned short*)(ws + off); off += 128 * 128 * 2;
    unsigned short* w2l_p = (unsigned short*)(ws + off); off += 128 * 128 * 2;
    unsigned short* w2r_p = (unsigned short*)(ws + off); off += 128 * 128 * 2;
    unsigned short* w3l_p = (unsigned short*)(ws + off); off += 128 * 128 * 2;
    unsigned short* w3r_p = (unsigned short*)(ws + off); off += 128 * 128 * 2;
    unsigned short* wlin_p = (unsigned short*)(ws + off); off += 128 * 64 * 2;

    const int TB = 256;
    auto blocks = [](long n, int tb) { return (unsigned)((n + tb - 1) / tb); };

    // 1) zero all accumulators (contiguous region: sums + counts)
    long nz = (long)N_TAXON * D_FEAT + N_TAXON + (long)N_SOTU * D_FEAT + N_SOTU;
    k_zero<<<blocks(nz, TB), TB, 0, stream>>>(sum_taxon, nz);

    // 2) bf16 feature copies
    k_f32_to_bf16<<<blocks((long)N_TAXON * D_FEAT, TB), TB, 0, stream>>>(x_taxon, xt_bf, (long)N_TAXON * D_FEAT);
    k_f32_to_bf16<<<blocks((long)N_SOTU * D_FEAT, TB), TB, 0, stream>>>(x_sotu, xs_bf, (long)N_SOTU * D_FEAT);

    // 3) pack weights into WMMA B-fragment order
    int totD = (128 / 32) * (128 / 16) * 512;      // 16384
    int totL = (128 / 32) * (64 / 16) * 512;       // 8192
    k_pack_w<<<blocks(totD, TB), TB, 0, stream>>>(W1l, w1l_p, 128, totD);
    k_pack_w<<<blocks(totD, TB), TB, 0, stream>>>(W1r, w1r_p, 128, totD);
    k_pack_w<<<blocks(totD, TB), TB, 0, stream>>>(W2l, w2l_p, 128, totD);
    k_pack_w<<<blocks(totD, TB), TB, 0, stream>>>(W2r, w2r_p, 128, totD);
    k_pack_w<<<blocks(totD, TB), TB, 0, stream>>>(W3l, w3l_p, 128, totD);
    k_pack_w<<<blocks(totD, TB), TB, 0, stream>>>(W3r, w3r_p, 128, totD);
    k_pack_w<<<blocks(totL, TB), TB, 0, stream>>>(Wlin, wlin_p, 64, totL);

    // 4) conv1: taxon -> taxon
    k_scatter<<<blocks((long)E_HP * 32, TB), TB, 0, stream>>>(xt_bf, hp_src, hp_dst, sum_taxon, cnt_taxon, E_HP, 1);
    k_mean_bf16<<<blocks((long)N_TAXON * D_FEAT, TB), TB, 0, stream>>>(sum_taxon, cnt_taxon, mean_t, N_TAXON);
    k_sage_gemm<<<blocks((long)(N_TAXON / 16) * 32, TB), TB, 0, stream>>>(mean_t, xt_bf, w1l_p, w1r_p, b1, taxon_h, N_TAXON);

    // 5) conv2: (taxon, sotu)
    k_scatter<<<blocks((long)E_RH * 32, TB), TB, 0, stream>>>(xt_bf, rh_src, rh_dst, sum_sotu, cnt_sotu, E_RH, 1);
    k_mean_bf16<<<blocks((long)N_SOTU * D_FEAT, TB), TB, 0, stream>>>(sum_sotu, cnt_sotu, mean_s, N_SOTU);
    k_sage_gemm<<<blocks((long)(N_SOTU / 16) * 32, TB), TB, 0, stream>>>(mean_s, xs_bf, w2l_p, w2r_p, b2, sotu_h, N_SOTU);

    // 6) conv3: (taxon_h, sotu_h)  — counts unchanged, re-zero sums only
    long nz2 = (long)N_SOTU * D_FEAT;
    k_zero<<<blocks(nz2, TB), TB, 0, stream>>>(sum_sotu, nz2);
    k_scatter<<<blocks((long)E_RH * 32, TB), TB, 0, stream>>>(taxon_h, rh_src, rh_dst, sum_sotu, cnt_sotu, E_RH, 0);
    k_mean_bf16<<<blocks((long)N_SOTU * D_FEAT, TB), TB, 0, stream>>>(sum_sotu, cnt_sotu, mean_s, N_SOTU);
    k_sage_gemm<<<blocks((long)(N_SOTU / 16) * 32, TB), TB, 0, stream>>>(mean_s, sotu_h, w3l_p, w3r_p, b3, sotu_h, N_SOTU);

    // 7) final linear -> f32 output
    k_final<<<blocks((long)(N_SOTU / 16) * 32, TB), TB, 0, stream>>>(sotu_h, wlin_p, blin, outp, N_SOTU);
}
